// CNSYN_59528246723247
// MI455X (gfx1250) — matrix operand order repaired and checked by reading, not compiled, not run on hardware
//
#include <hip/hip_runtime.h>
#include <hip/hip_bf16.h>

// ---------------------------------------------------------------------------
// Problem constants (match reference)
// ---------------------------------------------------------------------------
#define Bn     512
#define Sn     64
#define Cn     10
#define En     100
#define Ep     128          // E padded to K-multiple of 32
#define NH     1024
#define CH     2048
#define CH2    1024
#define SETROWS   (Bn*Sn)          // 32768
#define MROWS     (SETROWS + Bn)   // 33280 = 128*260  (set rows + inst rows)

typedef __attribute__((ext_vector_type(16))) __bf16 v16bf;
typedef __attribute__((ext_vector_type(8)))  float  v8f;

union FragAB { v16bf v; unsigned int u[8]; };
union FragC  { v8f   v; float        f[8]; };

// ---------------------------------------------------------------------------
// gfx1250 async global->LDS staging (ASYNCcnt-tracked DMA, no VGPR round-trip)
// ---------------------------------------------------------------------------
__device__ __forceinline__ unsigned lds_off_u(const void* p) {
    return (unsigned)(size_t)(__attribute__((address_space(3))) const void*)p;
}
__device__ __forceinline__ void g2l_async_b128(unsigned lds, const void* g) {
    asm volatile("global_load_async_to_lds_b128 %0, %1, off"
                 :: "v"(lds), "v"(g) : "memory");
}
// async loads complete in order: waiting to <=4 outstanding means the
// previous tile's 4 copies have landed while the next tile's 4 are in flight.
__device__ __forceinline__ void wait_async_le4() {
    asm volatile("s_wait_asynccnt 0x4" ::: "memory");
}
__device__ __forceinline__ void wait_async0() {
    asm volatile("s_wait_asynccnt 0x0" ::: "memory");
}

// ---------------------------------------------------------------------------
// Convert f32 -> bf16 TRANSPOSED with zero padding:
//   dst[n][k] = src[k][n]   (weights become [N x Kp] row-major = "Bt")
// so GEMM staging is a pure byte copy (async b128) and all LDS ops are b128.
// ---------------------------------------------------------------------------
__global__ void k_cvt_pad_t(const float* __restrict__ src, int srows, int scols,
                            __bf16* __restrict__ dst, int dN, int dK)
{
    int idx = blockIdx.x * blockDim.x + threadIdx.x;
    int total = dN * dK;
    if (idx >= total) return;
    int n = idx / dK, k = idx - n * dK;
    float v = (k < srows && n < scols) ? src[(size_t)k * scols + n] : 0.0f;
    dst[idx] = (__bf16)v;
}

// ---------------------------------------------------------------------------
// Build X for the embedding path: rows 0..32767 = emb[set_ids], rows
// 32768..33279 = emb[inst_ids]; zero-pad cols 100..127; also write mask.
// ---------------------------------------------------------------------------
__global__ void k_fill_x(const int* __restrict__ set_ids,
                         const int* __restrict__ inst_ids,
                         const float* __restrict__ emb,
                         __bf16* __restrict__ X,
                         float* __restrict__ maskf)
{
    int idx = blockIdx.x * blockDim.x + threadIdx.x;      // MROWS*128 threads
    int r = idx >> 7, e = idx & 127;
    if (r >= MROWS) return;
    int id; float mk;
    if (r < SETROWS) { id = set_ids[r]; mk = (id != 0) ? 1.0f : 0.0f; }
    else             { id = inst_ids[r - SETROWS]; mk = 1.0f; }
    float v = (e < En) ? emb[(size_t)id * En + e] : 0.0f;
    X[(size_t)r * Ep + e] = (__bf16)v;
    if (e == 0) maskf[r] = mk;
}

// ---------------------------------------------------------------------------
// Context aggregation: one wave32 per row.  scores_i = <ctx_i, ent>,
// alpha = scores / sum(scores), out = sum_i alpha_i * ctx_i.
// NaN->0 masking applied only for set rows (matches reference).
// ---------------------------------------------------------------------------
__global__ __launch_bounds__(256)
void k_ctx(const int* __restrict__ set_ids,
           const int* __restrict__ inst_ids,
           const int* __restrict__ ctxarr,
           const float* __restrict__ emb,
           __bf16* __restrict__ X)
{
    int wrow = blockIdx.x * (blockDim.x >> 5) + (threadIdx.x >> 5);
    int lane = threadIdx.x & 31;
    if (wrow >= MROWS) return;
    bool nan_mask = (wrow < SETROWS);
    int id = nan_mask ? set_ids[wrow] : inst_ids[wrow - SETROWS];

    // this lane covers dims {lane, lane+32, lane+64, lane+96}
    float ent[4];
#pragma unroll
    for (int j = 0; j < 4; ++j) {
        int e = lane + 32 * j;
        ent[j] = (e < En) ? emb[(size_t)id * En + e] : 0.0f;
    }
    float cval[Cn][4];
    float score[Cn];
    float total = 0.0f;
#pragma unroll
    for (int i = 0; i < Cn; ++i) {
        int cid = ctxarr[(size_t)id * Cn + i];
        float p = 0.0f;
#pragma unroll
        for (int j = 0; j < 4; ++j) {
            int e = lane + 32 * j;
            float cv = (e < En) ? emb[(size_t)cid * En + e] : 0.0f;
            cval[i][j] = cv;
            p += cv * ent[j];
        }
        for (int off = 16; off; off >>= 1) p += __shfl_xor(p, off, 32);
        score[i] = p;
        total += p;
    }
    float out[4] = {0.f, 0.f, 0.f, 0.f};
#pragma unroll
    for (int i = 0; i < Cn; ++i) {
        float al = score[i] / total;
#pragma unroll
        for (int j = 0; j < 4; ++j) out[j] += al * cval[i][j];
    }
#pragma unroll
    for (int j = 0; j < 4; ++j) {
        int e = lane + 32 * j;                 // always < 128
        float v = (e < En) ? out[j] : 0.0f;
        if (nan_mask && v != v) v = 0.0f;
        X[(size_t)wrow * Ep + e] = (__bf16)v;
    }
}

// ---------------------------------------------------------------------------
// bf16 WMMA GEMM:  C[M,N] = act(A[M,K] @ Bt[N,K]^T + bias)
//   A  [M x K]  bf16 row-major (lda = K)
//   Bt [N x K]  bf16 row-major (pre-transposed weights, ldb = K)
// Requirements: M%128==0, N%128==0, K%32==0.
// Block = 256 threads = 8 waves (4M x 2N); block tile 128x128;
// wave tile 32x64 -> 8 WMMAs per K-step per wave.
// Double-buffered GLOBAL_LOAD_ASYNC_TO_LDS_B128 pipeline hides DMA latency
// behind the matrix pipe; all LDS reads are ds_load_b128.
// ---------------------------------------------------------------------------
__global__ __launch_bounds__(256)
void k_gemm(const __bf16* __restrict__ A, int lda,
            const __bf16* __restrict__ Bt, int ldb,
            const float* __restrict__ bias,
            __bf16* __restrict__ C, int ldc,
            int K, int relu)
{
    __shared__ __align__(16) unsigned short As[2][128][32];   // [buf][m][k]
    __shared__ __align__(16) unsigned short Bs[2][128][32];   // [buf][n][k]

    const int tid   = threadIdx.x;
    const int lane  = tid & 31;
    const int wid   = tid >> 5;
    const int waveM = wid >> 1;          // 0..3  -> 32-row slab
    const int waveN = wid & 1;           // 0..1  -> 64-col half
    const int row0  = blockIdx.x * 128;
    const int col0  = blockIdx.y * 128;

    FragC acc[2][4];
#pragma unroll
    for (int s = 0; s < 2; ++s)
#pragma unroll
        for (int t = 0; t < 4; ++t)
#pragma unroll
            for (int i = 0; i < 8; ++i) acc[s][t].f[i] = 0.0f;

    // staging: each thread owns a 16-element (32B) strip = two b128 copies
    const int sRow = tid >> 1;           // 0..127
    const int sCol = (tid & 1) << 4;     // 0,16

    unsigned ldsA[2], ldsB[2];
    ldsA[0] = lds_off_u(&As[0][sRow][sCol]);
    ldsA[1] = lds_off_u(&As[1][sRow][sCol]);
    ldsB[0] = lds_off_u(&Bs[0][sRow][sCol]);
    ldsB[1] = lds_off_u(&Bs[1][sRow][sCol]);

    const __bf16* ap = A  + (size_t)(row0 + sRow) * lda + sCol;
    const __bf16* bp = Bt + (size_t)(col0 + sRow) * ldb + sCol;

    // prologue: stage tile 0 into buffer 0
    g2l_async_b128(ldsA[0],      ap);
    g2l_async_b128(ldsA[0] + 16, ap + 8);
    g2l_async_b128(ldsB[0],      bp);
    g2l_async_b128(ldsB[0] + 16, bp + 8);

    const int lan15  = lane & 15;
    const int kBaseA = (lane < 16) ? 0 : 8;
    const int kBaseB = (lane < 16) ? 0 : 8;

    int cur = 0;
    for (int kk = 0; kk < K; kk += 32) {
        const int nxt = cur ^ 1;
        if (kk + 32 < K) {
            // issue next tile's DMA into the other buffer, then drain only
            // the current tile (in-order completion).
            const __bf16* an = ap + kk + 32;
            const __bf16* bn = bp + kk + 32;
            g2l_async_b128(ldsA[nxt],      an);
            g2l_async_b128(ldsA[nxt] + 16, an + 8);
            g2l_async_b128(ldsB[nxt],      bn);
            g2l_async_b128(ldsB[nxt] + 16, bn + 8);
            wait_async_le4();
        } else {
            wait_async0();
        }
        __syncthreads();

        // ---- fragments per documented 16-bit WMMA layouts ----
        // A 16x32: lanes 0-15 M=lane, K {0..7,16..23}; lanes 16-31 {8..15,24..31}
        FragAB a[2];
#pragma unroll
        for (int s = 0; s < 2; ++s) {
            const int mrow = (waveM << 5) + (s << 4) + lan15;
#pragma unroll
            for (int r = 0; r < 8; ++r) {
                int k = kBaseA + ((r < 4) ? (r << 1) : (16 + ((r - 4) << 1)));
                a[s].u[r] = *(const unsigned int*)&As[cur][mrow][k];
            }
        }
        // B 32x16: lanes 0-15 K {0..7,16..23}, lanes 16-31 K {8..15,24..31}; N = lane%16
        FragAB b[4];
#pragma unroll
        for (int t = 0; t < 4; ++t) {
            const int nrow = (waveN << 6) + (t << 4) + lan15;
#pragma unroll
            for (int r = 0; r < 8; ++r)
                b[t].u[r] = *(const unsigned int*)&Bs[cur][nrow][kBaseB + ((r < 4) ? (r << 1) : (16 + ((r - 4) << 1)))];
        }
#pragma unroll
        for (int s = 0; s < 2; ++s)
#pragma unroll
            for (int t = 0; t < 4; ++t)
                acc[s][t].v = __builtin_amdgcn_wmma_f32_16x16x32_bf16(
                    false, a[s].v, false, b[t].v, (short)0, acc[s][t].v, false, false);
        __syncthreads();   // readers done before buffer 'cur' is re-filled
        cur = nxt;
    }

    // D layout: VGPR r -> M = r + (lane<16 ? 0 : 8), N = lane%16
    const int mB = row0 + (waveM << 5) + ((lane < 16) ? 0 : 8);
#pragma unroll
    for (int s = 0; s < 2; ++s)
#pragma unroll
        for (int t = 0; t < 4; ++t) {
            const int n = col0 + (waveN << 6) + (t << 4) + lan15;
            const float bv = bias ? bias[n] : 0.0f;
#pragma unroll
            for (int r = 0; r < 8; ++r) {
                float x = acc[s][t].f[r] + bv;
                if (relu) x = fmaxf(x, 0.0f);
                C[(size_t)(mB + (s << 4) + r) * ldc + n] = (__bf16)x;
            }
        }
}

// ---------------------------------------------------------------------------
// Masked reduction over S: Eset[b][n] = sum_s H2[b*64+s][n]*mask[b*64+s]
// ---------------------------------------------------------------------------
__global__ void k_reduce(const __bf16* __restrict__ H2,
                         const float* __restrict__ maskf,
                         float* __restrict__ Eset)
{
    int idx = blockIdx.x * blockDim.x + threadIdx.x;   // 512*1024
    if (idx >= Bn * NH) return;
    int b = idx >> 10, n = idx & (NH - 1);
    float s = 0.0f;
#pragma unroll 4
    for (int j = 0; j < Sn; ++j) {
        int r = b * Sn + j;
        s += (float)H2[(size_t)r * NH + n] * maskf[r];
    }
    Eset[idx] = s;
}

// ---------------------------------------------------------------------------
// Build q2 input: rows 0..511 = Eset, rows 512..1023 = Eset + instEmbed
// (instEmbed = H2 rows 32768..33279, no mask — matches reference)
// ---------------------------------------------------------------------------
__global__ void k_build_y(const float* __restrict__ Eset,
                          const __bf16* __restrict__ H2,
                          __bf16* __restrict__ Y)
{
    int idx = blockIdx.x * blockDim.x + threadIdx.x;   // 1024*1024
    if (idx >= 2 * Bn * NH) return;
    int r = idx >> 10, n = idx & (NH - 1);
    float v;
    if (r < Bn) v = Eset[(size_t)r * NH + n];
    else        v = Eset[(size_t)(r - Bn) * NH + n]
                  + (float)H2[(size_t)(SETROWS + (r - Bn)) * NH + n];
    Y[idx] = (__bf16)v;
}

// ---------------------------------------------------------------------------
// Final row-dot:  out[r] = dot(G2[r,:], w3) + b3   (one wave per row)
// ---------------------------------------------------------------------------
__global__ __launch_bounds__(256)
void k_rowdot(const __bf16* __restrict__ G2,
              const float* __restrict__ w3,
              const float* __restrict__ b3,
              float* __restrict__ out, int rows)
{
    int w = blockIdx.x * (blockDim.x >> 5) + (threadIdx.x >> 5);
    int lane = threadIdx.x & 31;
    if (w >= rows) return;
    float s = 0.0f;
    for (int n = lane; n < CH2; n += 32)
        s += (float)G2[(size_t)w * CH2 + n] * w3[n];
    for (int off = 16; off; off >>= 1) s += __shfl_xor(s, off, 32);
    if (lane == 0) out[w] = s + b3[0];
}

// ---------------------------------------------------------------------------
// Host side
// ---------------------------------------------------------------------------
static inline size_t align256(size_t x) { return (x + 255) & ~(size_t)255; }

extern "C" void kernel_launch(void* const* d_in, const int* in_sizes, int n_in,
                              void* d_out, int out_size, void* d_ws, size_t ws_size,
                              hipStream_t stream)
{
    (void)in_sizes; (void)n_in; (void)out_size; (void)ws_size;
    const int*   set_ids  = (const int*)  d_in[0];
    const int*   inst_ids = (const int*)  d_in[1];
    const int*   ctxarr   = (const int*)  d_in[2];
    const float* emb      = (const float*)d_in[3];

    // weight sets: [w1, w2, b2, q2w1, q2b1, q2w2, q2b2, q2w3, q2b3]
    const float* Wreg[9] = { (const float*)d_in[4],  (const float*)d_in[5],
                             (const float*)d_in[6],  (const float*)d_in[7],
                             (const float*)d_in[8],  (const float*)d_in[9],
                             (const float*)d_in[10], (const float*)d_in[11],
                             (const float*)d_in[12] };
    const float* What[9] = { (const float*)d_in[13], (const float*)d_in[14],
                             (const float*)d_in[15], (const float*)d_in[16],
                             (const float*)d_in[17], (const float*)d_in[18],
                             (const float*)d_in[19], (const float*)d_in[20],
                             (const float*)d_in[21] };

    // ---- workspace carve-up ----
    char* ws = (char*)d_ws;
    size_t o = 0;
    auto carve = [&](size_t bytes) { char* p = ws + o; o = align256(o + bytes); return p; };
    __bf16* w1t   = (__bf16*)carve((size_t)Ep * Ep * 2);    // [128 x 128]  (N x K)
    __bf16* w2t   = (__bf16*)carve((size_t)NH * Ep * 2);    // [1024 x 128]
    __bf16* q2w1t = (__bf16*)carve((size_t)CH * NH * 2);    // [2048 x 1024]
    __bf16* q2w2t = (__bf16*)carve((size_t)CH2 * CH * 2);   // [1024 x 2048]
    __bf16* X     = (__bf16*)carve((size_t)MROWS * Ep * 2);
    __bf16* H1    = (__bf16*)carve((size_t)MROWS * Ep * 2);
    __bf16* H2    = (__bf16*)carve((size_t)MROWS * NH * 2);
    float*  Eset  = (float*) carve((size_t)Bn * NH * 4);
    __bf16* Y     = (__bf16*)carve((size_t)2 * Bn * NH * 2);
    __bf16* G1    = (__bf16*)carve((size_t)2 * Bn * CH * 2);
    __bf16* G2    = (__bf16*)carve((size_t)2 * Bn * CH2 * 2);
    float*  maskf = (float*) carve((size_t)MROWS * 4);

    float* out = (float*)d_out;

    auto cvt_t = [&](const float* s, int sr, int sc, __bf16* d, int dN, int dK) {
        int total = dN * dK;
        k_cvt_pad_t<<<(total + 255) / 256, 256, 0, stream>>>(s, sr, sc, d, dN, dK);
    };

    for (int chain = 0; chain < 2; ++chain) {
        const float* const* W = chain ? What : Wreg;

        // weights -> bf16, transposed to [N x K] so GEMM staging is byte-copy
        cvt_t(W[0], En, En,  w1t,   Ep,  Ep);
        cvt_t(W[1], En, NH,  w2t,   NH,  Ep);
        cvt_t(W[3], NH, CH,  q2w1t, CH,  NH);
        cvt_t(W[5], CH, CH2, q2w2t, CH2, CH);

        // build X (and mask on chain 0)
        if (chain == 0) {
            int total = MROWS * Ep;
            k_fill_x<<<(total + 255) / 256, 256, 0, stream>>>(set_ids, inst_ids, emb, X, maskf);
        } else {
            k_ctx<<<MROWS / 8, 256, 0, stream>>>(set_ids, inst_ids, ctxarr, emb, X);
        }

        // q1: H1 = relu(X @ W1) ; H2 = relu(H1 @ W2 + b2)
        k_gemm<<<dim3(MROWS / 128, Ep / 128), 256, 0, stream>>>(
            X, Ep, w1t, Ep, (const float*)nullptr, H1, Ep, Ep, 1);
        k_gemm<<<dim3(MROWS / 128, NH / 128), 256, 0, stream>>>(
            H1, Ep, w2t, Ep, W[2], H2, NH, Ep, 1);

        // masked sum over S, then build the two q2 input batches
        k_reduce<<<(Bn * NH + 255) / 256, 256, 0, stream>>>(H2, maskf, Eset);
        k_build_y<<<(2 * Bn * NH + 255) / 256, 256, 0, stream>>>(Eset, H2, Y);

        // q2: G1 = relu(Y @ q2w1 + b1) ; G2 = relu(G1 @ q2w2 + b2) ; out = G2 @ w3 + b3
        k_gemm<<<dim3(2 * Bn / 128, CH / 128), 256, 0, stream>>>(
            Y, NH, q2w1t, NH, W[4], G1, CH, NH, 1);
        k_gemm<<<dim3(2 * Bn / 128, CH2 / 128), 256, 0, stream>>>(
            G1, CH, q2w2t, CH, W[6], G2, CH2, CH, 1);
        k_rowdot<<<(2 * Bn) / 8, 256, 0, stream>>>(G2, W[7], W[8], out + chain * 2 * Bn, 2 * Bn);
    }
}
